// A16Matmul_180388626619
// MI455X (gfx1250) — matrix operand order repaired and checked by reading, not compiled, not run on hardware
//
#include <hip/hip_runtime.h>

typedef __bf16 bf16;
typedef __attribute__((ext_vector_type(16))) __bf16 v16bf;
typedef __attribute__((ext_vector_type(8)))  __bf16 v8bf;
typedef __attribute__((ext_vector_type(4)))  __bf16 v4bf;
typedef __attribute__((ext_vector_type(8)))  float  v8f;

#define K_DIM 4096
#define N_DIM 11008
#define M_DIM 256
#define GS    128      // quant group size along K
#define BK    32       // one bf16-WMMA K step
#define BM    128
#define BN    128
#define LDA   56       // padded LDS row stride (bf16 elems): 112B = 16B-aligned, 28-bank step

__global__ __launch_bounds__(256)
void gptq_w4a32_bf16wmma(const float* __restrict__ A,
                         const int*   __restrict__ Wq,
                         const float* __restrict__ Sc,
                         const float* __restrict__ Zp,
                         const float* __restrict__ Bias,
                         float*       __restrict__ Out)
{
    // double-buffered ping-pong tiles: one barrier per K-step
    __shared__ __align__(16) bf16 Alds[2][BM * LDA];   // row-major [m][k]
    __shared__ __align__(16) bf16 Wlds[2][BN * LDA];   // n-major  [n][k]

    const int t    = threadIdx.x;
    const int lane = t & 31;
    const int half = lane >> 4;   // K-half selector within a fragment
    const int l16  = lane & 15;
    const int wave = t >> 5;      // 0..7
    const int wm   = wave >> 2;   // 0..1 : 64-row slab
    const int wn   = wave & 3;    // 0..3 : 32-col slab

    const int m0 = blockIdx.y * BM;
    const int n0 = blockIdx.x * BN;

    // A staging: thread covers rows (t>>3)+32p, k-quad (t&7)
    const int ar  = t >> 3;       // 0..31
    const int akq = t & 7;        // 0..7  -> k = akq*4
    // W staging: thread owns column n = (t&127), k rows wk0..wk0+15
    const int wcol = t & 127;
    const int wk0  = (t >> 7) << 4;   // 0 or 16

    const v8f vzero = {0.f,0.f,0.f,0.f,0.f,0.f,0.f,0.f};
    v8f acc[4][2];
    #pragma unroll
    for (int mt = 0; mt < 4; ++mt)
        #pragma unroll
        for (int nt = 0; nt < 2; ++nt) acc[mt][nt] = vzero;

    // pipeline registers (tile in flight)
    float4 av[4];
    int    q[16];
    float  sc, zp;

    // ---- issue global loads for K-step ks into pipeline registers ----
    auto load_tile = [&](int ks) {
        const int k0  = ks * BK;
        const int g   = k0 >> 7;
        const int kin = (k0 & (GS - 1)) + wk0;
        #pragma unroll
        for (int p = 0; p < 4; ++p)
            av[p] = *(const float4*)(A + (size_t)(m0 + ar + p * 32) * K_DIM + k0 + akq * 4);
        const int* wp = Wq + ((size_t)g * GS + kin) * N_DIM + (n0 + wcol);
        #pragma unroll
        for (int i = 0; i < 16; ++i) q[i] = wp[(size_t)i * N_DIM];
        sc = Sc[(size_t)g * N_DIM + n0 + wcol];
        zp = Zp[(size_t)g * N_DIM + n0 + wcol];
    };

    // ---- speculative prefetch of K-step ks weight rows (global_prefetch_b8) ----
    auto prefetch_tile = [&](int ks) {
        const int k0  = ks * BK;
        const int g   = k0 >> 7;
        const int kin = (k0 & (GS - 1)) + wk0;
        const int* wp = Wq + ((size_t)g * GS + kin) * N_DIM + (n0 + wcol);
        #pragma unroll
        for (int i = 0; i < 16; i += 4)
            __builtin_prefetch(wp + (size_t)i * N_DIM, 0, 1);
    };

    // ---- convert + stage pipeline registers into LDS buffer `buf` ----
    auto store_tile = [&](int buf) {
        #pragma unroll
        for (int p = 0; p < 4; ++p) {
            v4bf a4 = { (bf16)av[p].x, (bf16)av[p].y, (bf16)av[p].z, (bf16)av[p].w };
            *(v4bf*)&Alds[buf][(ar + p * 32) * LDA + akq * 4] = a4;
        }
        union { v8bf h[2]; bf16 e[16]; } wb;
        #pragma unroll
        for (int i = 0; i < 16; ++i)
            wb.e[i] = (bf16)(((float)q[i] - zp) * sc);
        *(v8bf*)&Wlds[buf][wcol * LDA + wk0]     = wb.h[0];
        *(v8bf*)&Wlds[buf][wcol * LDA + wk0 + 8] = wb.h[1];
    };

    // ---- fragment loads + 8 WMMAs from LDS buffer `buf` ----
    auto compute_tile = [&](int buf) {
        v16bf af[4], bfr[2];
        #pragma unroll
        for (int mt = 0; mt < 4; ++mt) {
            const bf16* rp = &Alds[buf][(wm * 64 + mt * 16 + l16) * LDA];
            union { v16bf v; v8bf h[2]; } u;
            u.h[0] = *(const v8bf*)(rp + half * 16);        // K {0..7} or {16..23}
            u.h[1] = *(const v8bf*)(rp + half * 16 + 8);    // K {8..15} or {24..31}
            af[mt] = u.v;
        }
        #pragma unroll
        for (int nt = 0; nt < 2; ++nt) {
            const bf16* cp = &Wlds[buf][(wn * 32 + nt * 16 + l16) * LDA + half * 16];
            union { v16bf v; v8bf h[2]; } u;
            u.h[0] = *(const v8bf*)(cp);
            u.h[1] = *(const v8bf*)(cp + 8);
            bfr[nt] = u.v;
        }
        #pragma unroll
        for (int mt = 0; mt < 4; ++mt)
            #pragma unroll
            for (int nt = 0; nt < 2; ++nt)
                acc[mt][nt] = __builtin_amdgcn_wmma_f32_16x16x32_bf16(
                    false, af[mt], false, bfr[nt], (short)0, acc[mt][nt], false, false);
    };

    const int KSTEPS = K_DIM / BK;   // 128

    // prologue: stage tile 0
    load_tile(0);
    store_tile(0);
    prefetch_tile(1);
    __syncthreads();

    for (int ks = 0; ks < KSTEPS; ++ks) {
        const int cur  = ks & 1;
        const bool more = (ks + 1 < KSTEPS);

        if (more) {
            load_tile(ks + 1);               // global loads in flight during compute
            if (ks + 2 < KSTEPS) prefetch_tile(ks + 2);
        }

        compute_tile(cur);                   // ds_loads + WMMAs overlap load latency

        if (more) store_tile(1 - cur);       // waits loadcnt only after WMMAs issued

        __syncthreads();                     // single barrier: publish next, retire cur
    }

    // ---- epilogue: bias add + store (C layout: VGPR j -> M=j(+8 hi half), N=l16) ----
    #pragma unroll
    for (int nt = 0; nt < 2; ++nt) {
        const int col = n0 + wn * 32 + nt * 16 + l16;
        const float b = Bias[col];
        #pragma unroll
        for (int mt = 0; mt < 4; ++mt) {
            const int rowb = m0 + wm * 64 + mt * 16 + half * 8;
            #pragma unroll
            for (int j = 0; j < 8; ++j)
                Out[(size_t)(rowb + j) * N_DIM + col] = acc[mt][nt][j] + b;
        }
    }
}

extern "C" void kernel_launch(void* const* d_in, const int* in_sizes, int n_in,
                              void* d_out, int out_size, void* d_ws, size_t ws_size,
                              hipStream_t stream) {
    const float* A    = (const float*)d_in[0];
    const int*   Wq   = (const int*)d_in[1];
    const float* Sc   = (const float*)d_in[2];
    const float* Zp   = (const float*)d_in[3];
    const float* Bias = (const float*)d_in[4];
    float*       Out  = (float*)d_out;

    dim3 grid(N_DIM / BN, M_DIM / BM);   // 86 x 2
    gptq_w4a32_bf16wmma<<<grid, 256, 0, stream>>>(A, Wq, Sc, Zp, Bias, Out);

    (void)in_sizes; (void)n_in; (void)out_size; (void)d_ws; (void)ws_size;
}
